// RNNAcceptor_24567212934028
// MI455X (gfx1250) — matrix operand order repaired and checked by reading, not compiled, not run on hardware
//
#include <hip/hip_runtime.h>
#include <hip/hip_bf16.h>

// ---------------- problem dims ----------------
#define B_   256
#define T_   512
#define VV_  32000
#define E_   512
#define H_   1024
#define MM_  1024
#define O_   2
#define G4H  (4*H_)      // 4096
#define KCAT (E_ + H_)   // 1536

typedef __bf16 bf16_t;
typedef __attribute__((ext_vector_type(16))) __bf16 v16bf;
typedef __attribute__((ext_vector_type(8)))  __bf16 v8bf;
typedef __attribute__((ext_vector_type(8)))  float   v8f;

// A-tile staging in LDS: 64 rows x 32 k (bf16), row stride padded to 40 elems
// (80B, 16B-aligned, 20 dwords -> 16 distinct banks across a 16-lane group).
#define AROW 40

// ---------------- WMMA fragment loaders (ISA 7.12.2 layouts, wave32) ------
// A 16x32 bf16 (MxK): lane L -> row M = L%16, half h = L/16.
//   element e<8  : k = 8h + e ; element e>=8 : k = 8h + 8 + e
__device__ inline v16bf pack_frag(v8bf lo, v8bf hi) {
    v16bf r;
#pragma unroll
    for (int i = 0; i < 8; ++i) { r[i] = lo[i]; r[i + 8] = hi[i]; }
    return r;
}
__device__ inline v16bf load_a_frag(const bf16_t* __restrict__ row, int kbase, int h) {
    return pack_frag(*(const v8bf*)(row + kbase + 8 * h),
                     *(const v8bf*)(row + kbase + 8 * h + 16));
}
// B 32x16 bf16 (KxN): lane L -> col N = L%16, half h = L/16, e<8: k = 8h + e ; e>=8: k = 8h + 8 + e.
// Weights stored row-major [N,K] (== B^T): two 16B chunks per lane (k offsets 8h and 8h + 16).
__device__ inline v16bf load_b_frag(const bf16_t* __restrict__ wrow, int kbase, int h) {
    return pack_frag(*(const v8bf*)(wrow + kbase + 8 * h),
                     *(const v8bf*)(wrow + kbase + 8 * h + 16));
}

// async global->LDS copy of one 16B chunk (CDNA5, ASYNCcnt-tracked)
__device__ inline void async_copy_b128(unsigned lds_addr, const void* gptr) {
    unsigned long long ga = (unsigned long long)(size_t)gptr;
    asm volatile("global_load_async_to_lds_b128 %0, %1, off"
                 :: "v"(lds_addr), "v"(ga) : "memory");
}
__device__ inline void wait_async_all() {
    asm volatile("s_wait_asynccnt 0x0" ::: "memory");
}

// ---------------- prep kernels ----------------
__global__ void cvt_f32_bf16_kernel(const float* __restrict__ src,
                                    bf16_t* __restrict__ dst, int n) {
    for (int i = blockIdx.x * blockDim.x + threadIdx.x; i < n;
         i += gridDim.x * blockDim.x)
        dst[i] = (bf16_t)src[i];
}

__global__ void build_wcat_kernel(const float* __restrict__ wih,
                                  const float* __restrict__ whh,
                                  bf16_t* __restrict__ wcat) {
    int n = G4H * KCAT;
    for (int i = blockIdx.x * blockDim.x + threadIdx.x; i < n;
         i += gridDim.x * blockDim.x) {
        int row = i / KCAT, k = i - row * KCAT;
        float v = (k < E_) ? wih[row * E_ + k] : whh[row * H_ + (k - E_)];
        wcat[i] = (bf16_t)v;
    }
}

__global__ void build_bias_kernel(const float* __restrict__ bih,
                                  const float* __restrict__ bhh,
                                  float* __restrict__ bias) {
    int i = blockIdx.x * blockDim.x + threadIdx.x;
    if (i < G4H) bias[i] = bih[i] + bhh[i];
}

// ---------------- fused LSTM step ------------------------------------------
// Block tile: 64 batch rows x 32 hidden cols x 4 gates.
// 8 waves: wave = (gate<<1)|msub ; each wave one 32x32 WMMA tile (2x2 frags).
// A ([emb_t | h_prev], K=1536) staged in LDS via async copies, double-buffered.
// Gate accumulators exchanged through LDS; cell update fused in-kernel.
__global__ __launch_bounds__(256)
void lstm_step_fused_kernel(const int* __restrict__ x,
                            const bf16_t* __restrict__ ebf,
                            const bf16_t* __restrict__ hprev,
                            bf16_t* __restrict__ hnext,
                            const bf16_t* __restrict__ wcat,
                            const float* __restrict__ bias,
                            float* __restrict__ c,
                            bf16_t* __restrict__ hnb,
                            const int* __restrict__ x_index,
                            int t) {
    __shared__ __align__(16) bf16_t lds_a[2][64 * AROW];   // 2 x 5120 B
    __shared__ float lds_g[4 * 64 * 32];                   // 32 KB

    const int tid  = threadIdx.x;
    const int lane = tid & 31;
    const int wave = tid >> 5;
    const int h    = lane >> 4;
    const int ln   = lane & 15;
    const int mw   = wave & 1;      // m subtile (0/1)
    const int g    = wave >> 1;     // gate 0..3 (i,f,g,o)
    const int m0   = blockIdx.y * 64;
    const int j0   = blockIdx.x * 32;

    // --- staging assignment: thread -> (row 0..63, 16B chunk 0..3) ---
    const int srow = tid >> 2;
    const int schk = tid & 3;
    const int tok  = x[(m0 + srow) * T_ + t];
    const bf16_t* se = ebf + (size_t)tok * E_;                 // k <  E_
    const bf16_t* sh = hprev + (size_t)(m0 + srow) * H_;       // k >= E_
    const unsigned lds_dst0 =
        (unsigned)(size_t)(&lds_a[0][srow * AROW]) + (unsigned)(schk * 16);
    const unsigned lds_dst1 =
        (unsigned)(size_t)(&lds_a[1][srow * AROW]) + (unsigned)(schk * 16);

    // --- B row pointers: weight rows n = g*H + j0 + nt*16 + ln ---
    const bf16_t* brow[2];
#pragma unroll
    for (int nt = 0; nt < 2; ++nt)
        brow[nt] = wcat + (size_t)(g * H_ + j0 + nt * 16 + ln) * KCAT;

    v8f acc[2][2] = {};

    // prefetch first k-chunk
    async_copy_b128(lds_dst0, se + schk * 8);

    for (int kk = 0; kk < KCAT; kk += 32) {
        const int cur = (kk >> 5) & 1;
        wait_async_all();
        __syncthreads();                       // chunk `cur` resident in LDS
        // prefetch next chunk into the other buffer
        const int kn = kk + 32;
        if (kn < KCAT) {
            const bf16_t* src = (kn < E_) ? (se + kn) : (sh + (kn - E_));
            async_copy_b128(cur ? lds_dst0 : lds_dst1, src + schk * 8);
        }
        // A fragments from LDS, B fragments from global (L2-resident weights)
        v16bf a[2], b[2];
#pragma unroll
        for (int mt = 0; mt < 2; ++mt) {
            const bf16_t* ap = &lds_a[cur][(mw * 32 + mt * 16 + ln) * AROW];
            a[mt] = pack_frag(*(const v8bf*)(ap + 8 * h),
                              *(const v8bf*)(ap + 8 * h + 16));
        }
#pragma unroll
        for (int nt = 0; nt < 2; ++nt)
            b[nt] = load_b_frag(brow[nt], kk, h);
#pragma unroll
        for (int mt = 0; mt < 2; ++mt)
#pragma unroll
            for (int nt = 0; nt < 2; ++nt)
                acc[mt][nt] = __builtin_amdgcn_wmma_f32_16x16x32_bf16(
                    false, a[mt], false, b[nt], (short)0, acc[mt][nt], false, false);
        __syncthreads();                       // done reading `cur` before overwrite
    }

    // --- exchange gate pre-activations through LDS ---
    // D layout: vgpr r, lane l -> M = r + 8*(l/16), N = l%16
#pragma unroll
    for (int mt = 0; mt < 2; ++mt)
#pragma unroll
        for (int nt = 0; nt < 2; ++nt)
#pragma unroll
            for (int r = 0; r < 8; ++r) {
                int ml = mw * 32 + mt * 16 + 8 * h + r;
                int jl = nt * 16 + ln;
                lds_g[(g * 64 + ml) * 32 + jl] = acc[mt][nt][r];
            }
    __syncthreads();

    // --- fused cell update: 64x32 cells, 8 per thread ---
    const int tcap = x_index ? 0 : 0; (void)tcap;
#pragma unroll
    for (int i = tid; i < 64 * 32; i += 256) {
        int ml = i >> 5, jl = i & 31;
        int m = m0 + ml, j = j0 + jl;
        float vi = lds_g[(0 * 64 + ml) * 32 + jl] + bias[j];
        float vf = lds_g[(1 * 64 + ml) * 32 + jl] + bias[H_ + j];
        float vg = lds_g[(2 * 64 + ml) * 32 + jl] + bias[2 * H_ + j];
        float vo = lds_g[(3 * 64 + ml) * 32 + jl] + bias[3 * H_ + j];
        vi = 1.f / (1.f + __expf(-vi));
        vf = 1.f / (1.f + __expf(-vf));
        vg = tanhf(vg);
        vo = 1.f / (1.f + __expf(-vo));
        size_t ci = (size_t)m * H_ + j;
        float cn = vf * c[ci] + vi * vg;
        c[ci] = cn;
        float hh = vo * tanhf(cn);
        hnext[ci] = (bf16_t)hh;
        if (x_index[m] == t) hnb[ci] = (bf16_t)hh;
    }
}

// ---------------- fc1 GEMM: z = tanh(hn @ fc1_w^T + b1) --------------------
__global__ __launch_bounds__(256)
void fc1_gemm_kernel(const bf16_t* __restrict__ hnb,
                     const bf16_t* __restrict__ w1,
                     const float* __restrict__ b1,
                     float* __restrict__ z) {
    const int lane = threadIdx.x & 31;
    const int wave = threadIdx.x >> 5;
    const int h    = lane >> 4;
    const int ln   = lane & 15;
    const int m0   = blockIdx.y * 64  + (wave & 1) * 32;
    const int n0   = blockIdx.x * 128 + (wave >> 1) * 32;

    const bf16_t* arow[2];
#pragma unroll
    for (int mt = 0; mt < 2; ++mt)
        arow[mt] = hnb + (size_t)(m0 + mt * 16 + ln) * H_;
    const bf16_t* brow[2];
#pragma unroll
    for (int nt = 0; nt < 2; ++nt)
        brow[nt] = w1 + (size_t)(n0 + nt * 16 + ln) * H_;

    v8f acc[2][2] = {};
    for (int kk = 0; kk < H_; kk += 32) {
        v16bf a[2], b[2];
#pragma unroll
        for (int mt = 0; mt < 2; ++mt) a[mt] = load_a_frag(arow[mt], kk, h);
#pragma unroll
        for (int nt = 0; nt < 2; ++nt) b[nt] = load_b_frag(brow[nt], kk, h);
#pragma unroll
        for (int mt = 0; mt < 2; ++mt)
#pragma unroll
            for (int nt = 0; nt < 2; ++nt)
                acc[mt][nt] = __builtin_amdgcn_wmma_f32_16x16x32_bf16(
                    false, a[mt], false, b[nt], (short)0, acc[mt][nt], false, false);
    }
#pragma unroll
    for (int mt = 0; mt < 2; ++mt)
#pragma unroll
        for (int nt = 0; nt < 2; ++nt)
#pragma unroll
            for (int r = 0; r < 8; ++r) {
                int m = m0 + mt * 16 + 8 * h + r;
                int n = n0 + nt * 16 + ln;
                z[(size_t)m * MM_ + n] = tanhf(acc[mt][nt][r] + b1[n]);
            }
}

// ---------------- fc2 + log_softmax (O=2, tiny) ----------------------------
__global__ void fc2_lsm_kernel(const float* __restrict__ z,
                               const float* __restrict__ w2,
                               const float* __restrict__ b2,
                               float* __restrict__ out) {
    int b = blockIdx.x * blockDim.x + threadIdx.x;
    if (b >= B_) return;
    const float* zr = z + (size_t)b * MM_;
    float a0 = b2[0], a1 = b2[1];
    for (int k = 0; k < MM_; ++k) {
        float zv = zr[k];
        a0 += zv * w2[k];
        a1 += zv * w2[MM_ + k];
    }
    float mx  = fmaxf(a0, a1);
    float lse = mx + logf(__expf(a0 - mx) + __expf(a1 - mx));
    out[b * 2 + 0] = a0 - lse;
    out[b * 2 + 1] = a1 - lse;
}

// ---------------- host side ----------------
extern "C" void kernel_launch(void* const* d_in, const int* in_sizes, int n_in,
                              void* d_out, int out_size, void* d_ws, size_t ws_size,
                              hipStream_t stream) {
    const int*   x       = (const int*)  d_in[0];
    const int*   x_index = (const int*)  d_in[1];
    const float* emb     = (const float*)d_in[2];
    const float* W_ih    = (const float*)d_in[3];
    const float* W_hh    = (const float*)d_in[4];
    const float* b_ih    = (const float*)d_in[5];
    const float* b_hh    = (const float*)d_in[6];
    const float* fc1_w   = (const float*)d_in[7];
    const float* fc1_b   = (const float*)d_in[8];
    const float* fc2_w   = (const float*)d_in[9];
    const float* fc2_b   = (const float*)d_in[10];
    const float* h0      = (const float*)d_in[11];
    const float* c0      = (const float*)d_in[12];
    float* out = (float*)d_out;

    // workspace layout (256B aligned slabs)
    char* ws = (char*)d_ws;
    size_t cur = 0;
    auto alloc = [&](size_t bytes) {
        char* p = ws + cur;
        cur += (bytes + 255) & ~(size_t)255;
        return p;
    };
    const int n_emb = (VV_ + 1) * E_;
    bf16_t* ebf  = (bf16_t*)alloc((size_t)n_emb * 2);
    bf16_t* wcat = (bf16_t*)alloc((size_t)G4H * KCAT * 2);
    bf16_t* w1b  = (bf16_t*)alloc((size_t)MM_ * H_ * 2);
    float*  bias = (float*) alloc((size_t)G4H * 4);
    bf16_t* hb0  = (bf16_t*)alloc((size_t)B_ * H_ * 2);
    bf16_t* hb1  = (bf16_t*)alloc((size_t)B_ * H_ * 2);
    float*  c    = (float*) alloc((size_t)B_ * H_ * 4);
    bf16_t* hnb  = (bf16_t*)alloc((size_t)B_ * H_ * 2);
    float*  z    = (float*) alloc((size_t)B_ * MM_ * 4);
    (void)ws_size; (void)in_sizes; (void)n_in; (void)out_size;

    // ---- prep: bf16 conversions + fused bias + state init ----
    cvt_f32_bf16_kernel<<<(n_emb + 255) / 256, 256, 0, stream>>>(emb, ebf, n_emb);
    build_wcat_kernel<<<(G4H * KCAT + 255) / 256, 256, 0, stream>>>(W_ih, W_hh, wcat);
    cvt_f32_bf16_kernel<<<(MM_ * H_ + 255) / 256, 256, 0, stream>>>(fc1_w, w1b, MM_ * H_);
    build_bias_kernel<<<(G4H + 255) / 256, 256, 0, stream>>>(b_ih, b_hh, bias);
    cvt_f32_bf16_kernel<<<(B_ * H_ + 255) / 256, 256, 0, stream>>>(h0, hb0, B_ * H_);
    hipMemcpyAsync(c, c0, (size_t)B_ * H_ * 4, hipMemcpyDeviceToDevice, stream);

    // ---- LSTM recurrence: 512 fused steps, h ping-pong ----
    dim3 step_grid(H_ / 32, B_ / 64);            // (32, 4)
    for (int t = 0; t < T_; ++t) {
        const bf16_t* hp = (t & 1) ? hb1 : hb0;
        bf16_t*       hn = (t & 1) ? hb0 : hb1;
        lstm_step_fused_kernel<<<step_grid, 256, 0, stream>>>(
            x, ebf, hp, hn, wcat, bias, c, hnb, x_index, t);
    }

    // ---- head ----
    dim3 fc1_grid(MM_ / 128, B_ / 64);           // (8, 4)
    fc1_gemm_kernel<<<fc1_grid, 256, 0, stream>>>(hnb, w1b, fc1_b, z);
    fc2_lsm_kernel<<<1, 256, 0, stream>>>(z, fc2_w, fc2_b, out);
}
